// ExpertWrappedBlock_13056700580557
// MI455X (gfx1250) — matrix operand order repaired and checked by reading, not compiled, not run on hardware
//
#include <hip/hip_runtime.h>
#include <hip/hip_bf16.h>

// ---------------------------------------------------------------- types
typedef __bf16 bf16_t;
typedef __attribute__((ext_vector_type(8)))  __bf16 v8bf;
typedef __attribute__((ext_vector_type(16))) __bf16 v16bf;
typedef __attribute__((ext_vector_type(8)))  float  v8f;

#define DEVI static __device__ __forceinline__

DEVI v8bf ld8(const bf16_t* p) { return *(const v8bf*)p; }
DEVI void st8(bf16_t* p, v8bf v) { *(v8bf*)p = v; }
DEVI v16bf cat16(v8bf a, v8bf b) {
  return __builtin_shufflevector(a, b, 0,1,2,3,4,5,6,7,8,9,10,11,12,13,14,15);
}
DEVI v8f wmma_bf16(v16bf a, v16bf b, v8f c) {
  return __builtin_amdgcn_wmma_f32_16x16x32_bf16(false, a, false, b, (short)0, c, false, false);
}

// CDNA5 async global->LDS copy (ASYNCcnt-tracked, no VGPR round trip)
DEVI void async_ld_b128(unsigned int lds_off, const void* gaddr) {
  asm volatile("global_load_async_to_lds_b128 %0, %1, off"
               :: "v"(lds_off), "v"(gaddr) : "memory");
}
DEVI void wait_async0() { asm volatile("s_wait_asynccnt 0" ::: "memory"); }
DEVI void wait_async4() { asm volatile("s_wait_asynccnt 4" ::: "memory"); }

// Problem constants (match reference)
#define Bz   8
#define Tz   832
#define Dz   1024
#define Hz   16
#define HDz  64
#define HIDz 4096
#define NBz  576
#define MTOT (Bz*Tz)          // 6656
#define NGRP 16               // (T-NB)/M expert groups per batch

// ---------------------------------------------------------------- weight transpose + f32->bf16
// W [K,N] f32 -> Wt [N,K] bf16   (32x32 LDS tile, block 32x8)
__global__ void transpose_convert_kernel(const float* __restrict__ W,
                                         bf16_t* __restrict__ Wt, int K, int N) {
  __shared__ float tile[32][33];
  int kT = blockIdx.y * 32, nT = blockIdx.x * 32;
  int tx = threadIdx.x, ty = threadIdx.y;
  #pragma unroll
  for (int i = 0; i < 32; i += 8)
    tile[ty + i][tx] = W[(size_t)(kT + ty + i) * N + nT + tx];
  __syncthreads();
  #pragma unroll
  for (int i = 0; i < 32; i += 8)
    Wt[(size_t)(nT + ty + i) * K + kT + tx] = (bf16_t)tile[tx][ty + i];
}

// lora_A [E,D,R=8] f32 -> At [E,16,D] bf16 (rows 8..15 zero)
__global__ void lora_a_convert_kernel(const float* __restrict__ A, bf16_t* __restrict__ At) {
  int i = blockIdx.x * 256 + threadIdx.x;        // e*16*D total
  int d = i & (Dz - 1);
  int r = (i >> 10) & 15;
  int e = i >> 14;
  float v = (r < 8) ? A[((size_t)e * Dz + d) * 8 + r] : 0.0f;
  At[i] = (bf16_t)v;
}

// lora_B [E,R=8,D] f32 -> Bt [E,D,8] bf16
__global__ void lora_b_convert_kernel(const float* __restrict__ Bm, bf16_t* __restrict__ Bt) {
  int i = blockIdx.x * 256 + threadIdx.x;        // e*D*8 total
  int r = i & 7;
  int d = (i >> 3) & (Dz - 1);
  int e = i >> 13;
  Bt[i] = (bf16_t)Bm[((size_t)e * 8 + r) * Dz + d];
}

// ---------------------------------------------------------------- LayerNorm (f32 in -> bf16 out)
__global__ __launch_bounds__(256)
void ln_kernel(const float* __restrict__ x, const float* __restrict__ g,
               const float* __restrict__ b, bf16_t* __restrict__ y) {
  int row = blockIdx.x;
  const float* xr = x + (size_t)row * Dz;
  __shared__ float redS[8], redQ[8];
  float s = 0.f, q = 0.f;
  for (int i = threadIdx.x; i < Dz; i += 256) { float v = xr[i]; s += v; q += v * v; }
  #pragma unroll
  for (int m = 16; m >= 1; m >>= 1) { s += __shfl_xor(s, m, 32); q += __shfl_xor(q, m, 32); }
  int wid = threadIdx.x >> 5;
  if ((threadIdx.x & 31) == 0) { redS[wid] = s; redQ[wid] = q; }
  __syncthreads();
  float ts = 0.f, tq = 0.f;
  #pragma unroll
  for (int w = 0; w < 8; ++w) { ts += redS[w]; tq += redQ[w]; }
  float mean = ts * (1.0f / Dz);
  float var  = tq * (1.0f / Dz) - mean * mean;
  float rstd = rsqrtf(var + 1e-6f);
  for (int i = threadIdx.x; i < Dz; i += 256) {
    float v = (xr[i] - mean) * rstd * g[i] + b[i];
    y[(size_t)row * Dz + i] = (bf16_t)v;
  }
}

// ---------------------------------------------------------------- generic bf16 WMMA GEMM
// C[M,N] = epilogue(A[M,K] @ Bt[N,K]^T + bias [+ res])
// 256 threads / 8 waves; 128x128 block tile; K-step 32; double-buffered
// async global->LDS staging (ASYNCcnt) overlapped with WMMA compute.
enum { EPI_F32_BIAS = 0, EPI_F32_BIAS_RES = 1, EPI_BF16_GELU = 2 };

__global__ __launch_bounds__(256)
void gemm_bf16_kernel(const bf16_t* __restrict__ A, const bf16_t* __restrict__ Bt,
                      const float* __restrict__ bias, const float* __restrict__ res,
                      float* __restrict__ outF, bf16_t* __restrict__ outB,
                      int M, int N, int K, int mode) {
  __shared__ __attribute__((aligned(16))) bf16_t lA[2][128 * 40];
  __shared__ __attribute__((aligned(16))) bf16_t lB[2][128 * 40];
  const int tid   = threadIdx.x;
  const int mBase = blockIdx.y * 128;
  const int nBase = blockIdx.x * 128;
  const int wid   = tid >> 5, lane = tid & 31;
  const int half  = lane >> 4, l16 = lane & 15;
  const int wr    = (wid >> 2) * 64;   // wave row base in tile
  const int wc    = (wid & 3) * 32;    // wave col base in tile

  // per-thread staging coordinates (two b128 slots each for A and B)
  const int r0 = tid >> 2,          c0 = (tid & 3) * 8;
  const int r1 = (tid + 256) >> 2,  c1 = ((tid + 256) & 3) * 8;
  unsigned int aoff[2], boff[2];
  aoff[0] = (unsigned int)(uintptr_t)&lA[0][0];
  aoff[1] = (unsigned int)(uintptr_t)&lA[1][0];
  boff[0] = (unsigned int)(uintptr_t)&lB[0][0];
  boff[1] = (unsigned int)(uintptr_t)&lB[1][0];

  auto issue_stage = [&](int buf, int kc) {
    async_ld_b128(aoff[buf] + (unsigned int)(r0 * 40 + c0) * 2,
                  &A[(size_t)(mBase + r0) * K + kc + c0]);
    async_ld_b128(boff[buf] + (unsigned int)(r0 * 40 + c0) * 2,
                  &Bt[(size_t)(nBase + r0) * K + kc + c0]);
    async_ld_b128(aoff[buf] + (unsigned int)(r1 * 40 + c1) * 2,
                  &A[(size_t)(mBase + r1) * K + kc + c1]);
    async_ld_b128(boff[buf] + (unsigned int)(r1 * 40 + c1) * 2,
                  &Bt[(size_t)(nBase + r1) * K + kc + c1]);
  };

  v8f acc[4][2];
  #pragma unroll
  for (int i = 0; i < 4; ++i)
    #pragma unroll
    for (int j = 0; j < 2; ++j) acc[i][j] = (v8f)0.0f;

  const int nK = K >> 5;
  issue_stage(0, 0);
  for (int kk = 0; kk < nK; ++kk) {
    const int cur = kk & 1;
    const bool more = (kk + 1) < nK;
    if (more) {
      issue_stage(cur ^ 1, (kk + 1) << 5);   // prefetch next stage
      wait_async4();                          // wait for current stage only
    } else {
      wait_async0();
    }
    __syncthreads();
    v16bf af[4], bfr[2];
    #pragma unroll
    for (int i = 0; i < 4; ++i) {
      const bf16_t* p = &lA[cur][(wr + i * 16 + l16) * 40 + half * 8];
      af[i] = cat16(ld8(p), ld8(p + 16));                 // A-frag: k = half*8+e / 16+half*8+e
    }
    #pragma unroll
    for (int j = 0; j < 2; ++j) {
      const bf16_t* p = &lB[cur][(wc + j * 16 + l16) * 40 + half * 8];
      bfr[j] = cat16(ld8(p), ld8(p + 16));                // B-frag: k = half*8+e / 16+half*8+e
    }
    #pragma unroll
    for (int i = 0; i < 4; ++i)
      #pragma unroll
      for (int j = 0; j < 2; ++j)
        acc[i][j] = wmma_bf16(af[i], bfr[j], acc[i][j]);
    __syncthreads();   // protect buffer reuse before next prefetch lands
  }

  // epilogue: lane holds column n = ...+l16, rows m = ...+half*8+e
  #pragma unroll
  for (int j = 0; j < 2; ++j) {
    int n = nBase + wc + j * 16 + l16;
    float bv = bias[n];
    #pragma unroll
    for (int i = 0; i < 4; ++i) {
      int m0 = mBase + wr + i * 16 + half * 8;
      #pragma unroll
      for (int e = 0; e < 8; ++e) {
        int m = m0 + e;
        float v = acc[i][j][e] + bv;
        if (mode == EPI_BF16_GELU) {
          v = 0.5f * v * (1.0f + erff(v * 0.70710678118f));
          outB[(size_t)m * N + n] = (bf16_t)v;
        } else {
          if (mode == EPI_F32_BIAS_RES) v += res[(size_t)m * N + n];
          outF[(size_t)m * N + n] = v;
        }
      }
    }
  }
}

// ---------------------------------------------------------------- QKV split/scale/transpose
// qkv f32 [B,T,3,H,hd] -> q bf16 [B,H,T,hd] (pre-scaled), k bf16 [B,H,T,hd], vT bf16 [B,H,hd,T]
__global__ void qkv_split_kernel(const float* __restrict__ qkv, bf16_t* __restrict__ qb,
                                 bf16_t* __restrict__ kb, bf16_t* __restrict__ vtb) {
  int i = blockIdx.x * 256 + threadIdx.x;       // B*T*H*hd
  int d = i & 63;
  int h = (i >> 6) & 15;
  int t = (i >> 10) % Tz;
  int b = i / (Dz * Tz);
  size_t base = (size_t)(b * Tz + t) * (3 * Dz) + h * HDz + d;
  float qv = qkv[base], kv = qkv[base + Dz], vv = qkv[base + 2 * Dz];
  size_t bh = (size_t)(b * Hz + h);
  qb[(bh * Tz + t) * HDz + d]  = (bf16_t)(qv * 0.125f);   // * hd^-0.5
  kb[(bh * Tz + t) * HDz + d]  = (bf16_t)kv;
  vtb[(bh * HDz + d) * Tz + t] = (bf16_t)vv;
}

// ---------------------------------------------------------------- flash attention
// block = 128 threads (4 waves), each wave owns 16 query rows; grid = B*H * (T/64)
__global__ __launch_bounds__(128)
void attn_kernel(const bf16_t* __restrict__ q, const bf16_t* __restrict__ k,
                 const bf16_t* __restrict__ vt, const float* __restrict__ mask,
                 bf16_t* __restrict__ o) {
  __shared__ __attribute__((aligned(16))) bf16_t lP[4][16 * 32];
  const int wid = threadIdx.x >> 5, lane = threadIdx.x & 31;
  const int half = lane >> 4, l16 = lane & 15;
  const int bh    = blockIdx.x / (Tz / 64);
  const int qBase = (blockIdx.x % (Tz / 64)) * 64 + wid * 16;

  const bf16_t* qrow = q + ((size_t)bh * Tz + qBase + l16) * HDz;
  v16bf qa0 = cat16(ld8(qrow + half * 8),      ld8(qrow + 16 + half * 8));
  v16bf qa1 = cat16(ld8(qrow + 32 + half * 8), ld8(qrow + 48 + half * 8));

  v8f O[4];
  #pragma unroll
  for (int c = 0; c < 4; ++c) O[c] = (v8f)0.0f;
  float mrow[8], lrow[8];
  #pragma unroll
  for (int j = 0; j < 8; ++j) { mrow[j] = -3.0e38f; lrow[j] = 0.0f; }

  for (int kc = 0; kc < Tz; kc += 32) {
    v8f s[2];
    #pragma unroll
    for (int st = 0; st < 2; ++st) {
      s[st] = (v8f)0.0f;
      const bf16_t* krow = k + ((size_t)bh * Tz + kc + st * 16 + l16) * HDz;
      v16bf kb0 = cat16(ld8(krow + half * 8),      ld8(krow + 16 + half * 8));
      v16bf kb1 = cat16(ld8(krow + 32 + half * 8), ld8(krow + 48 + half * 8));
      s[st] = wmma_bf16(qa0, kb0, s[st]);
      s[st] = wmma_bf16(qa1, kb1, s[st]);
      #pragma unroll
      for (int j = 0; j < 8; ++j)
        s[st][j] += mask[(size_t)(qBase + half * 8 + j) * Tz + kc + st * 16 + l16];
    }
    // online softmax, per lane tracks rows half*8+j (replicated over 16 lanes/half)
    #pragma unroll
    for (int j = 0; j < 8; ++j) {
      float v0 = s[0][j], v1 = s[1][j];
      float mx = fmaxf(v0, v1);
      #pragma unroll
      for (int dd = 8; dd >= 1; dd >>= 1) mx = fmaxf(mx, __shfl_xor(mx, dd, 16));
      float mn = fmaxf(mrow[j], mx);
      float al = __expf(mrow[j] - mn);
      float p0 = __expf(v0 - mn), p1 = __expf(v1 - mn);
      float rs = p0 + p1;
      #pragma unroll
      for (int dd = 8; dd >= 1; dd >>= 1) rs += __shfl_xor(rs, dd, 16);
      lrow[j] = lrow[j] * al + rs;
      mrow[j] = mn;
      #pragma unroll
      for (int c = 0; c < 4; ++c) O[c][j] *= al;
      lP[wid][(half * 8 + j) * 32 + l16]      = (bf16_t)p0;
      lP[wid][(half * 8 + j) * 32 + 16 + l16] = (bf16_t)p1;
    }
    __syncthreads();
    const bf16_t* prow = &lP[wid][l16 * 32];
    v16bf pa = cat16(ld8(prow + half * 8), ld8(prow + 16 + half * 8));
    #pragma unroll
    for (int c = 0; c < 4; ++c) {
      const bf16_t* vrow = vt + ((size_t)bh * HDz + c * 16 + l16) * Tz + kc;
      v16bf vb = cat16(ld8(vrow + half * 8), ld8(vrow + 16 + half * 8));
      O[c] = wmma_bf16(pa, vb, O[c]);
    }
    __syncthreads();
  }
  const int b = bh >> 4, h = bh & 15;
  #pragma unroll
  for (int j = 0; j < 8; ++j) {
    float inv = 1.0f / lrow[j];
    int t = qBase + half * 8 + j;
    #pragma unroll
    for (int c = 0; c < 4; ++c)
      o[(size_t)(b * Tz + t) * Dz + h * HDz + c * 16 + l16] = (bf16_t)(O[c][j] * inv);
  }
}

// ---------------------------------------------------------------- LoRA (per batch x expert group)
// down[16,8] = z[16,1024] @ A_e ; out += (down @ B_e) * 1/8   — one wave per (b,grp)
__global__ __launch_bounds__(32)
void lora_kernel(const bf16_t* __restrict__ snorm, const bf16_t* __restrict__ At,
                 const bf16_t* __restrict__ Bt, const int* __restrict__ act,
                 float* __restrict__ out) {
  __shared__ __attribute__((aligned(16))) bf16_t lD[16 * 32];
  const int lane = threadIdx.x, half = lane >> 4, l16 = lane & 15;
  const int b = blockIdx.x >> 4, grp = blockIdx.x & 15;
  const int e = act[grp];
  for (int i = lane; i < 512; i += 32) lD[i] = (bf16_t)0.0f;

  const int t0 = NBz + grp * 16;
  const bf16_t* arow  = snorm + (size_t)(b * Tz + t0 + l16) * Dz;
  const bf16_t* atrow = At + (size_t)(e * 16 + l16) * Dz;
  v8f dacc = (v8f)0.0f;
  for (int dc = 0; dc < Dz; dc += 32) {
    v16bf a  = cat16(ld8(arow + dc + half * 8),   ld8(arow + dc + 16 + half * 8));
    v16bf bb = cat16(ld8(atrow + dc + half * 8),  ld8(atrow + dc + 16 + half * 8));
    dacc = wmma_bf16(a, bb, dacc);
  }
  #pragma unroll
  for (int j = 0; j < 8; ++j)
    lD[(half * 8 + j) * 32 + l16] = (bf16_t)dacc[j];
  __syncthreads();
  const bf16_t* prow = &lD[l16 * 32];
  v16bf pa = cat16(ld8(prow + half * 8), ld8(prow + 16 + half * 8));  // k 8..31 are zero

  v8bf z8 = (v8bf)(__bf16)0.0f;
  for (int dt = 0; dt < Dz / 16; ++dt) {
    const bf16_t* brow = Bt + ((size_t)e * Dz + dt * 16 + l16) * 8;
    v8bf lo = z8;
    if (half == 0) lo = ld8(brow);          // k=0..7 valid, rest zero
    v16bf vb = cat16(lo, z8);
    v8f up = (v8f)0.0f;
    up = wmma_bf16(pa, vb, up);
    #pragma unroll
    for (int j = 0; j < 8; ++j) {
      int t = t0 + half * 8 + j;
      size_t idx = (size_t)(b * Tz + t) * Dz + dt * 16 + l16;
      out[idx] += up[j] * 0.125f;           // LORA_SCALE = 1/R
    }
  }
}

// ---------------------------------------------------------------- host launch
extern "C" void kernel_launch(void* const* d_in, const int* in_sizes, int n_in,
                              void* d_out, int out_size, void* d_ws, size_t ws_size,
                              hipStream_t stream) {
  const float* s       = (const float*)d_in[0];
  const float* mask    = (const float*)d_in[1];
  const float* g1      = (const float*)d_in[2];
  const float* b1      = (const float*)d_in[3];
  const float* w_qkv   = (const float*)d_in[4];
  const float* b_qkv   = (const float*)d_in[5];
  const float* w_proj  = (const float*)d_in[6];
  const float* b_proj  = (const float*)d_in[7];
  const float* g2      = (const float*)d_in[8];
  const float* b2      = (const float*)d_in[9];
  const float* w_fc1   = (const float*)d_in[10];
  const float* b_fc1   = (const float*)d_in[11];
  const float* w_fc2   = (const float*)d_in[12];
  const float* b_fc2   = (const float*)d_in[13];
  const float* lora_A  = (const float*)d_in[14];
  const float* lora_B  = (const float*)d_in[15];
  const int*   act     = (const int*)d_in[16];
  float* out = (float*)d_out;
  char* ws = (char*)d_ws;

  // workspace layout (bytes); liveness-based aliasing
  const size_t off_wqkvT = 0;                                   // 3072*1024*2
  const size_t off_wprojT = off_wqkvT + (size_t)3072*1024*2;    // 1024*1024*2
  const size_t off_wfc1T  = off_wprojT + (size_t)1024*1024*2;   // 4096*1024*2
  const size_t off_wfc2T  = off_wfc1T + (size_t)4096*1024*2;    // 1024*4096*2
  const size_t off_At     = off_wfc2T + (size_t)1024*4096*2;    // 16*16*1024*2
  const size_t off_Bt     = off_At + (size_t)16*16*1024*2;      // 16*1024*8*2
  const size_t off_xln    = off_Bt + (size_t)16*1024*8*2;       // MT*1024*2
  const size_t off_qkv    = off_xln + (size_t)MTOT*1024*2;      // MT*3072*4
  const size_t off_q      = off_qkv + (size_t)MTOT*3072*4;      // MT*1024*2
  const size_t off_k      = off_q + (size_t)MTOT*1024*2;        // MT*1024*2
  const size_t off_vt     = off_k + (size_t)MTOT*1024*2;        // MT*1024*2
  const size_t off_o      = off_vt + (size_t)MTOT*1024*2;       // MT*1024*2
  const size_t off_s2     = off_q;      // aliases q+k (dead after attention)
  const size_t off_snorm  = off_vt;     // aliases vT (dead after attention)
  const size_t off_h      = off_qkv;    // aliases qkv f32 (dead after split)

  bf16_t* wqkvT  = (bf16_t*)(ws + off_wqkvT);
  bf16_t* wprojT = (bf16_t*)(ws + off_wprojT);
  bf16_t* wfc1T  = (bf16_t*)(ws + off_wfc1T);
  bf16_t* wfc2T  = (bf16_t*)(ws + off_wfc2T);
  bf16_t* Atb    = (bf16_t*)(ws + off_At);
  bf16_t* Btb    = (bf16_t*)(ws + off_Bt);
  bf16_t* xln    = (bf16_t*)(ws + off_xln);
  float*  qkvf   = (float*)(ws + off_qkv);
  bf16_t* qb     = (bf16_t*)(ws + off_q);
  bf16_t* kb     = (bf16_t*)(ws + off_k);
  bf16_t* vtb    = (bf16_t*)(ws + off_vt);
  bf16_t* ob     = (bf16_t*)(ws + off_o);
  float*  s2     = (float*)(ws + off_s2);
  bf16_t* snorm  = (bf16_t*)(ws + off_snorm);
  bf16_t* hb     = (bf16_t*)(ws + off_h);

  dim3 tb(32, 8);
  transpose_convert_kernel<<<dim3(3072/32, 1024/32), tb, 0, stream>>>(w_qkv,  wqkvT, 1024, 3072);
  transpose_convert_kernel<<<dim3(1024/32, 1024/32), tb, 0, stream>>>(w_proj, wprojT, 1024, 1024);
  transpose_convert_kernel<<<dim3(4096/32, 1024/32), tb, 0, stream>>>(w_fc1,  wfc1T, 1024, 4096);
  transpose_convert_kernel<<<dim3(1024/32, 4096/32), tb, 0, stream>>>(w_fc2,  wfc2T, 4096, 1024);
  lora_a_convert_kernel<<<(16*16*1024)/256, 256, 0, stream>>>(lora_A, Atb);
  lora_b_convert_kernel<<<(16*1024*8)/256, 256, 0, stream>>>(lora_B, Btb);

  ln_kernel<<<MTOT, 256, 0, stream>>>(s, g1, b1, xln);

  gemm_bf16_kernel<<<dim3(3072/128, MTOT/128), 256, 0, stream>>>(
      xln, wqkvT, b_qkv, nullptr, qkvf, nullptr, MTOT, 3072, 1024, EPI_F32_BIAS);

  qkv_split_kernel<<<(Bz*Tz*Hz*HDz)/256, 256, 0, stream>>>(qkvf, qb, kb, vtb);

  attn_kernel<<<Bz*Hz*(Tz/64), 128, 0, stream>>>(qb, kb, vtb, mask, ob);

  gemm_bf16_kernel<<<dim3(1024/128, MTOT/128), 256, 0, stream>>>(
      ob, wprojT, b_proj, s, s2, nullptr, MTOT, 1024, 1024, EPI_F32_BIAS_RES);

  ln_kernel<<<MTOT, 256, 0, stream>>>(s2, g2, b2, snorm);

  gemm_bf16_kernel<<<dim3(4096/128, MTOT/128), 256, 0, stream>>>(
      snorm, wfc1T, b_fc1, nullptr, nullptr, hb, MTOT, 4096, 1024, EPI_BF16_GELU);

  gemm_bf16_kernel<<<dim3(1024/128, MTOT/128), 256, 0, stream>>>(
      hb, wfc2T, b_fc2, s2, out, nullptr, MTOT, 1024, 4096, EPI_F32_BIAS_RES);

  lora_kernel<<<Bz*NGRP, 32, 0, stream>>>(snorm, Atb, Btb, act, out);
}
